// Embedding_15367392985163
// MI455X (gfx1250) — matrix operand order repaired and checked by reading, not compiled, not run on hardware
//
#include <hip/hip_runtime.h>
#include <hip/hip_bf16.h>

typedef __attribute__((ext_vector_type(16))) _Float16 v16h;
typedef __attribute__((ext_vector_type(8)))  _Float16 v8h;
typedef __attribute__((ext_vector_type(8)))  float    v8f;

union H16 { v16h v; v8h h[2]; };

#define ZPAD 72   // padded LDS row stride in halves (144B, conflict-spreading, 16B aligned)

// ---------------------------------------------------------------------------
// Kernel 0: transpose + convert weights to f16 so WMMA B-fragments are
// contiguous 16B loads.  w2 (64x1024 f32) -> w2t (1024x64 f16),
//                        w1 (64x144  f32) -> w1t (144x64  f16)
// ---------------------------------------------------------------------------
__global__ void cvt_transpose_kernel(const float* __restrict__ w2,
                                     const float* __restrict__ w1,
                                     _Float16* __restrict__ w2t,
                                     _Float16* __restrict__ w1t) {
  int i = blockIdx.x * blockDim.x + threadIdx.x;
  if (i < 1024 * 64) {
    int c = i >> 6, r = i & 63;                 // w2t[c][r] = w2[r][c]
    w2t[i] = (_Float16)w2[r * 1024 + c];
  }
  int j = i - 1024 * 64;
  if (j >= 0 && j < 144 * 64) {
    int c = j >> 6, r = j & 63;                 // w1t[c][r] = w1[r][c]
    w1t[j] = (_Float16)w1[r * 144 + c];
  }
}

// ---------------------------------------------------------------------------
// Kernel 1: fused hypernet.  grid = 256 WGs (16 tiles each), block = 128 (4 waves)
// ---------------------------------------------------------------------------
__global__ __launch_bounds__(128)
void hyper_fused_kernel(const float* __restrict__ z,
                        const float* __restrict__ b2,
                        const float* __restrict__ b1,
                        const _Float16* __restrict__ w2t,
                        const _Float16* __restrict__ w1t,
                        float* __restrict__ out) {
  __shared__ __align__(16) _Float16 zb[16 * ZPAD];            // z block, f16
  __shared__ __align__(16) _Float16 albuf[16 * 16 * ZPAD];    // a: [tile][i][zc], f16

  const int tid  = threadIdx.x;
  const int lane = tid & 31;
  const int wave = tid >> 5;            // 0..3
  const int g    = blockIdx.x;          // tile group: tiles [16g, 16g+16)
  const int lrow = lane & 15;           // lane % 16
  const int lhi  = lane >> 4;           // lane / 16

  // ---- stage z block (16 rows x 64) into LDS as f16 ----
  for (int idx = tid; idx < 16 * 64; idx += 128) {
    int r = idx >> 6, c = idx & 63;
    zb[r * ZPAD + c] = (_Float16)z[(g * 16 + r) * 64 + c];
  }
  __syncthreads();

  // ---- phase 1: C1 = zb @ w2 + b2 ; this wave owns column-tiles [16w, 16w+16) ----
  // A fragments (16x32 f16), k-steps 0,1; shared across all column tiles.
  H16 a1[2];
  {
    const _Float16* base = &zb[lrow * ZPAD + 8 * lhi];
    a1[0].h[0] = *(const v8h*)(base + 0);
    a1[0].h[1] = *(const v8h*)(base + 16);
    a1[1].h[0] = *(const v8h*)(base + 32);
    a1[1].h[1] = *(const v8h*)(base + 48);
  }
  for (int cc = 0; cc < 16; ++cc) {
    const int c   = wave * 16 + cc;          // column tile 0..63
    const int col = c * 16 + lrow;           // this lane's C1 column
    // B fragments (32x16) from w2t: row = col, K in table order (8h+s, 16+8h+s)
    const _Float16* wb = &w2t[col * 64 + 8 * lhi];
    H16 bk0, bk1;
    bk0.h[0] = *(const v8h*)(wb + 0);
    bk0.h[1] = *(const v8h*)(wb + 16);
    bk1.h[0] = *(const v8h*)(wb + 32);
    bk1.h[1] = *(const v8h*)(wb + 48);
    v8f acc = {};
    acc = __builtin_amdgcn_wmma_f32_16x16x32_f16(false, a1[0].v, false, bk0.v,
                                                 (short)0, acc, false, false);
    acc = __builtin_amdgcn_wmma_f32_16x16x32_f16(false, a1[1].v, false, bk1.v,
                                                 (short)0, acc, false, false);
    const float bias = b2[col];
    const int i_in = c >> 2;                 // a's IN index = col/64
    const int zc   = (c & 3) * 16 + lrow;    // a's z index  = col%64
#pragma unroll
    for (int v = 0; v < 8; ++v) {
      const int m = v + 8 * lhi;             // C1 row = tile-local n
      albuf[(m * 16 + i_in) * ZPAD + zc] = (_Float16)(acc[v] + bias);
    }
  }
  __syncthreads();

  // ---- phase 2: per tile n: a_n (16x64) @ w1 (64x144) + b1 ; 4 tiles per wave ----
  for (int tt = 0; tt < 4; ++tt) {
    const int nl = wave * 4 + tt;            // tile-local 0..15
    const int n  = g * 16 + nl;
    const int h  = n >> 6, kk = n & 63;
    // A fragments from LDS a-buffer: row index = IN index (lane % 16)
    H16 a2[2];
    {
      const _Float16* ab = &albuf[(nl * 16 + lrow) * ZPAD + 8 * lhi];
      a2[0].h[0] = *(const v8h*)(ab + 0);
      a2[0].h[1] = *(const v8h*)(ab + 16);
      a2[1].h[0] = *(const v8h*)(ab + 32);
      a2[1].h[1] = *(const v8h*)(ab + 48);
    }
    float* orow = out + (size_t)(h * 16) * 9216 + (size_t)kk * 144;
    for (int c2 = 0; c2 < 9; ++c2) {
      const int col = c2 * 16 + lrow;        // 0..143
      const _Float16* wb = &w1t[col * 64 + 8 * lhi];
      H16 bk0, bk1;
      bk0.h[0] = *(const v8h*)(wb + 0);
      bk0.h[1] = *(const v8h*)(wb + 16);
      bk1.h[0] = *(const v8h*)(wb + 32);
      bk1.h[1] = *(const v8h*)(wb + 48);
      v8f acc = {};
      acc = __builtin_amdgcn_wmma_f32_16x16x32_f16(false, a2[0].v, false, bk0.v,
                                                   (short)0, acc, false, false);
      acc = __builtin_amdgcn_wmma_f32_16x16x32_f16(false, a2[1].v, false, bk1.v,
                                                   (short)0, acc, false, false);
      const float bias = b1[col];
      // torch-view remap collapses: out[(h*16 + row)*9216 + kk*144 + col]
      // -> rows give contiguous 16-float (64B) coalesced runs per half-wave.
#pragma unroll
      for (int v = 0; v < 8; ++v) {
        const int i = v + 8 * lhi;           // D row = OUT index after the view
        orow[(size_t)i * 9216 + col] = acc[v] + bias;
      }
    }
  }
}

// ---------------------------------------------------------------------------
extern "C" void kernel_launch(void* const* d_in, const int* in_sizes, int n_in,
                              void* d_out, int out_size, void* d_ws, size_t ws_size,
                              hipStream_t stream) {
  (void)in_sizes; (void)n_in; (void)out_size; (void)ws_size;
  const float* z  = (const float*)d_in[0];   // (4096, 64)
  const float* w2 = (const float*)d_in[1];   // (64, 1024)
  const float* b2 = (const float*)d_in[2];   // (1024,)
  const float* w1 = (const float*)d_in[3];   // (64, 144)
  const float* b1 = (const float*)d_in[4];   // (144,)
  float* out = (float*)d_out;                // (1024, 1024, 3, 3)

  _Float16* w2t = (_Float16*)d_ws;           // 1024*64 f16 = 128KB
  _Float16* w1t = w2t + 1024 * 64;           // 144*64  f16 = 18KB

  const int total = 1024 * 64 + 144 * 64;
  cvt_transpose_kernel<<<(total + 255) / 256, 256, 0, stream>>>(w2, w1, w2t, w1t);
  hyper_fused_kernel<<<256, 128, 0, stream>>>(z, b2, b1, w2t, w1t, out);
}
